// SwinTransformerBlock_78426102825240
// MI455X (gfx1250) — matrix operand order repaired and checked by reading, not compiled, not run on hardware
//
#include <hip/hip_runtime.h>
#include <math.h>

typedef _Float16 f16;
typedef _Float16 v16h __attribute__((ext_vector_type(16)));
typedef float    v8f  __attribute__((ext_vector_type(8)));

#define DIM    256
#define HEADS  8
#define HD     32
#define NPIX   12544          // 112*112
#define NTOK   50176          // 4*12544 == 1024*49
#define NWIN   1024
#define HID    1024
#define SCALE_QK 0.17677669529663687f   // 32^-0.5

// ---------------------------------------------------------------- utilities
__device__ __forceinline__ float wred_sum(float v) {
#pragma unroll
  for (int m = 1; m < 32; m <<= 1) v += __shfl_xor(v, m, 32);
  return v;
}

__device__ __forceinline__ void win_map(int grow, int& b, int& wi, int& tokr, int& tokc) {
  int win = grow / 49, n = grow - win * 49;
  b = win >> 8; wi = win & 255;
  int wh = wi >> 4, ww = wi & 15;
  int i = n / 7, j = n - i * 7;
  tokr = wh * 7 + i + 3; if (tokr >= 112) tokr -= 112;   // roll(-3) gather
  tokc = ww * 7 + j + 3; if (tokc >= 112) tokc -= 112;
}

// CDNA5 async global->LDS copy (16B per lane), tracked by ASYNCcnt.
// Low 32 bits of a generic pointer into the LDS aperture are the wave-relative
// LDS byte address (ISA 10.2 aperture mapping), which is what VDST supplies.
__device__ __forceinline__ void async_copy_b128(const f16* g, f16* s) {
  unsigned lds = (unsigned)(size_t)s;
  asm volatile("global_load_async_to_lds_b128 %0, %1, off"
               :: "v"(lds), "v"(g) : "memory");
}
template <int N>
__device__ __forceinline__ void wait_asynccnt() {
  asm volatile("s_wait_asynccnt %0" :: "n"(N) : "memory");
}

// ---------------------------------------------------------------- fp32->fp16
__global__ void cvt_f32_f16(const float* __restrict__ s, f16* __restrict__ d, int n) {
  int i = blockIdx.x * 256 + threadIdx.x;
  if (i < n) d[i] = (f16)s[i];
}

// --------------------------------------------- LN1 + shift + window partition
__global__ __launch_bounds__(256) void ln1_window_kernel(
    const float* __restrict__ x, const float* __restrict__ g, const float* __restrict__ be,
    const float* __restrict__ polar, f16* __restrict__ xw, float* __restrict__ pw)
{
  int grow = blockIdx.x * 8 + (threadIdx.x >> 5);
  int lane = threadIdx.x & 31;
  int b, wi, r, c; win_map(grow, b, wi, r, c);
  size_t tok = (size_t)b * NPIX + r * 112 + c;
  const float* xs = x + tok * DIM;
  float v[8], s = 0.f, ss = 0.f;
#pragma unroll
  for (int e = 0; e < 8; e++) { v[e] = xs[lane + 32 * e]; s += v[e]; ss += v[e] * v[e]; }
  s = wred_sum(s); ss = wred_sum(ss);
  float mean = s * (1.f / 256.f);
  float rstd = rsqrtf(ss * (1.f / 256.f) - mean * mean + 1e-5f);
  f16* d = xw + (size_t)grow * DIM;
#pragma unroll
  for (int e = 0; e < 8; e++) {
    int ch = lane + 32 * e;
    d[ch] = (f16)((v[e] - mean) * rstd * g[ch] + be[ch]);
  }
  if (lane < 2) pw[grow * 2 + lane] = polar[tok * 2 + lane];
}

// ---------------------------------------------------------------- LN2 (plain)
__global__ __launch_bounds__(256) void ln2_kernel(
    const float* __restrict__ x2, const float* __restrict__ g, const float* __restrict__ be,
    f16* __restrict__ yh)
{
  int t = blockIdx.x * 8 + (threadIdx.x >> 5);
  int lane = threadIdx.x & 31;
  const float* xs = x2 + (size_t)t * DIM;
  float v[8], s = 0.f, ss = 0.f;
#pragma unroll
  for (int e = 0; e < 8; e++) { v[e] = xs[lane + 32 * e]; s += v[e]; ss += v[e] * v[e]; }
  s = wred_sum(s); ss = wred_sum(ss);
  float mean = s * (1.f / 256.f);
  float rstd = rsqrtf(ss * (1.f / 256.f) - mean * mean + 1e-5f);
  f16* d = yh + (size_t)t * DIM;
#pragma unroll
  for (int e = 0; e < 8; e++) {
    int ch = lane + 32 * e;
    d[ch] = (f16)((v[e] - mean) * rstd * g[ch] + be[ch]);
  }
}

// ------------------------------------------------- vote MLP (3->16 relu ->256)
__global__ __launch_bounds__(256) void vote_window_kernel(
    const float* __restrict__ ve, const float* __restrict__ w1, const float* __restrict__ b1,
    const float* __restrict__ w2, const float* __restrict__ b2, f16* __restrict__ vlin)
{
  int grow = blockIdx.x * 8 + (threadIdx.x >> 5);
  int lane = threadIdx.x & 31;
  int b, wi, r, c; win_map(grow, b, wi, r, c);
  size_t tok = (size_t)b * NPIX + r * 112 + c;
  float e0 = ve[tok * 3 + 0], e1 = ve[tok * 3 + 1], e2 = ve[tok * 3 + 2];
  float h[16];
#pragma unroll
  for (int j = 0; j < 16; j++)
    h[j] = fmaxf(w1[j * 3] * e0 + w1[j * 3 + 1] * e1 + w1[j * 3 + 2] * e2 + b1[j], 0.f);
  f16* d = vlin + (size_t)grow * DIM;
#pragma unroll
  for (int e = 0; e < 8; e++) {
    int ch = lane + 32 * e;
    float o = b2[ch];
#pragma unroll
    for (int j = 0; j < 16; j++) o += w2[ch * 16 + j] * h[j];
    d[ch] = (f16)o;
  }
}

// ---------------------------------------------------------- tiled WMMA GEMM
// C[M,N] = A[M,K] @ W[N,K]^T  (weight rows == WMMA B columns, contiguous)
// Block tile 128x128, 8 waves x (32x64), double-buffered async global->LDS.
struct GemmP {
  const f16* A; const f16* W; const float* bias;
  int K; int mode;           // 0=qkv 1=proj 2=fc1 3=fc2
  const f16* vlin; f16* outh;
  const float* res; float* outf;
};

__global__ __launch_bounds__(256) void gemm_wmma_kernel(GemmP p)
{
  __shared__ __attribute__((aligned(16))) f16 As[2 * 128 * 40];
  __shared__ __attribute__((aligned(16))) f16 Bs[2 * 128 * 40];
  int tid = threadIdx.x, lane = tid & 31, w = tid >> 5;
  int wm = w >> 1, wn = w & 1;
  int m0 = blockIdx.x * 128, n0 = blockIdx.y * 128;
  int K = p.K;
  v8f acc[2][4];
  { v8f z = {};
#pragma unroll
    for (int mi = 0; mi < 2; mi++)
#pragma unroll
      for (int ni = 0; ni < 4; ni++) acc[mi][ni] = z; }

  // stage one 128x32 K-slab of A and W into LDS buffer `buf` (async, 4 ops/thread)
  auto stage = [&](int buf, int k0) {
#pragma unroll
    for (int i = 0; i < 2; i++) {
      int ch = tid * 2 + i, r = ch >> 2, o = (ch & 3) * 8;
      async_copy_b128(p.A + (size_t)(m0 + r) * K + k0 + o, As + buf * 5120 + r * 40 + o);
    }
#pragma unroll
    for (int i = 0; i < 2; i++) {
      int ch = tid * 2 + i, r = ch >> 2, o = (ch & 3) * 8;
      async_copy_b128(p.W + (size_t)(n0 + r) * K + k0 + o, Bs + buf * 5120 + r * 40 + o);
    }
  };

  int nk = K >> 5;
  stage(0, 0);
  for (int kb = 0; kb < nk; kb++) {
    int buf = kb & 1;
    if (kb + 1 < nk) { stage(buf ^ 1, (kb + 1) * 32); wait_asynccnt<4>(); }
    else             { wait_asynccnt<0>(); }
    __syncthreads();

    v16h af[2], bf[4];
#pragma unroll
    for (int mi = 0; mi < 2; mi++) {
      const f16* bp = As + buf * 5120 + (wm * 32 + mi * 16 + (lane & 15)) * 40
                      + ((lane < 16) ? 0 : 16);
#pragma unroll
      for (int t = 0; t < 16; t++) af[mi][t] = bp[t];
    }
#pragma unroll
    for (int ni = 0; ni < 4; ni++) {
      const f16* bp = Bs + buf * 5120 + (wn * 64 + ni * 16 + (lane & 15)) * 40
                      + ((lane < 16) ? 0 : 16);
#pragma unroll
      for (int t = 0; t < 16; t++) bf[ni][t] = bp[t];
    }
#pragma unroll
    for (int mi = 0; mi < 2; mi++)
#pragma unroll
      for (int ni = 0; ni < 4; ni++)
        acc[mi][ni] = __builtin_amdgcn_wmma_f32_16x16x32_f16(
            false, af[mi], false, bf[ni], (short)0, acc[mi][ni], false, false);
    __syncthreads();   // protects buf^1 reuse by next stage()
  }

#pragma unroll
  for (int mi = 0; mi < 2; mi++)
#pragma unroll
    for (int ni = 0; ni < 4; ni++)
#pragma unroll
      for (int e = 0; e < 8; e++) {
        int R = m0 + wm * 32 + mi * 16 + e + ((lane < 16) ? 0 : 8);
        int O = n0 + wn * 64 + ni * 16 + (lane & 15);
        float v = acc[mi][ni][e] + p.bias[O];
        if (p.mode == 0) {                     // qkv: add vote embedding
          int part = O >> 8, ch = O & 255;
          float vl = (float)p.vlin[(size_t)R * 256 + ch];
          v += (part == 2 ? 2.f : 1.f) * vl;   // v_eff = qkv2 + 2*vemb
          p.outh[(size_t)R * 768 + O] = (f16)v;
        } else if (p.mode == 1) {              // proj: window reverse + shortcut
          int b, wi, rr, cc; win_map(R, b, wi, rr, cc);
          size_t tok = (size_t)b * NPIX + rr * 112 + cc;
          p.outf[tok * 256 + O] = v + p.res[tok * 256 + O];
        } else if (p.mode == 2) {              // fc1: exact gelu
          v = 0.5f * v * (1.f + erff(v * 0.70710678118654752f));
          p.outh[(size_t)R * HID + O] = (f16)v;
        } else {                               // fc2: residual -> output
          p.outf[(size_t)R * 256 + O] = v + p.res[(size_t)R * 256 + O];
        }
      }
}

// ------------------------------------------------------------- attention core
__global__ __launch_bounds__(256) void attn_kernel(
    const f16* __restrict__ qkvh, const float* __restrict__ pw,
    const float* __restrict__ mask, const float* __restrict__ tau,
    const float* __restrict__ rw1, const float* __restrict__ rb1,
    const float* __restrict__ rw2, const float* __restrict__ rb2,
    f16* __restrict__ attn_out)
{
  __shared__ f16  s_rpe[8 * 64 * 64];   // rpe bias; reused for softmax probs
  __shared__ f16  s_vt[8 * 32 * 72];    // per-head V^T  [d][m]
  __shared__ float s_qn[8 * 64];
  __shared__ float s_kn[8 * 64];

  int win = blockIdx.x;
  int tid = threadIdx.x, lane = tid & 31, h = tid >> 5;
  size_t rowbase = (size_t)win * 49;

  // ---- cooperative: relative-position-bias MLP for all 8 heads
  for (int pidx = tid; pidx < 4096; pidx += 256) {
    int n = pidx >> 6, m = pidx & 63;
    float o[8];
#pragma unroll
    for (int hh = 0; hh < 8; hh++) o[hh] = 0.f;
    if (n < 49 && m < 49) {
      float d0 = pw[(rowbase + n) * 2 + 0] - pw[(rowbase + m) * 2 + 0];
      float d1 = pw[(rowbase + n) * 2 + 1] - pw[(rowbase + m) * 2 + 1];
#pragma unroll
      for (int hh = 0; hh < 8; hh++) o[hh] = rb2[hh];
#pragma unroll
      for (int j = 0; j < 16; j++) {
        float hv = fmaxf(rw1[j * 2] * d0 + rw1[j * 2 + 1] * d1 + rb1[j], 0.f);
#pragma unroll
        for (int hh = 0; hh < 8; hh++) o[hh] += rw2[hh * 16 + j] * hv;
      }
    }
#pragma unroll
    for (int hh = 0; hh < 8; hh++) s_rpe[hh * 4096 + pidx] = (f16)o[hh];
  }

  // ---- cooperative: q / k L2 norms
#pragma unroll
  for (int it = 0; it < 4; it++) {
    int task = tid + 256 * it;
    int hh = task >> 7, rem = task & 127, isK = rem >> 6, r = rem & 63;
    float sum = 0.f;
    if (r < 49) {
      const f16* pq = qkvh + (rowbase + r) * 768 + isK * 256 + hh * 32;
#pragma unroll
      for (int d = 0; d < 32; d++) { float f = (float)pq[d]; sum += f * f; }
    }
    float nv = sqrtf(sum);
    if (isK) s_kn[hh * 64 + r] = nv; else s_qn[hh * 64 + r] = nv;
  }

  // ---- per-wave: stage V_eff^T (v already holds qkv2 + 2*vemb)
  for (int m = 0; m < 64; m++) {
    f16 v = (f16)0.f;
    if (m < 49) v = qkvh[(rowbase + m) * 768 + 512 + h * 32 + lane];
    s_vt[h * 2304 + lane * 72 + m] = v;
  }
  __syncthreads();

  // ---- phase 1: S = q @ k^T  (64x64 per head, one K-step of 32)
  const f16* qb = qkvh + rowbase * 768 + h * 32;
  const f16* kb = qb + 256;
  v16h af[4], bf[4];
#pragma unroll
  for (int mt = 0; mt < 4; mt++) {
    int r = mt * 16 + (lane & 15);
    bool ok = r < 49;
    const f16* bp = qb + (size_t)r * 768 + ((lane < 16) ? 0 : 16);
#pragma unroll
    for (int t = 0; t < 16; t++) af[mt][t] = ok ? bp[t] : (f16)0.f;
  }
#pragma unroll
  for (int nt = 0; nt < 4; nt++) {
    int cix = nt * 16 + (lane & 15);
    bool ok = cix < 49;
    const f16* bp = kb + (size_t)cix * 768 + ((lane < 16) ? 0 : 16);
#pragma unroll
    for (int t = 0; t < 16; t++) bf[nt][t] = ok ? bp[t] : (f16)0.f;
  }
  v8f S[4][4];
  {
    v8f z = {};
#pragma unroll
    for (int mt = 0; mt < 4; mt++)
#pragma unroll
      for (int nt = 0; nt < 4; nt++)
        S[mt][nt] = __builtin_amdgcn_wmma_f32_16x16x32_f16(
            false, af[mt], false, bf[nt], (short)0, z, false, false);
  }

  // ---- epilogue: cosine norm, tau, rpe, shift-mask
  float taui = 1.f / fmaxf(tau[h], 0.01f);
  const float* mwin = mask + (size_t)(win & 255) * 2401;
#pragma unroll
  for (int mt = 0; mt < 4; mt++)
#pragma unroll
    for (int nt = 0; nt < 4; nt++)
#pragma unroll
      for (int e = 0; e < 8; e++) {
        int r = mt * 16 + e + ((lane < 16) ? 0 : 8);
        int c = nt * 16 + (lane & 15);
        float s = S[mt][nt][e] * SCALE_QK;
        s /= fmaxf(s_qn[h * 64 + r] * s_kn[h * 64 + c], 1e-6f);
        s *= taui;
        s += (float)s_rpe[h * 4096 + r * 64 + c];
        if (r < 49 && c < 49) s += mwin[r * 49 + c];
        if (c >= 49) s = -1e30f;
        S[mt][nt][e] = s;
      }

  // ---- softmax over rows (across nt tiles, then 16-lane half reductions)
#pragma unroll
  for (int mt = 0; mt < 4; mt++)
#pragma unroll
    for (int e = 0; e < 8; e++) {
      float mx = -1e30f;
#pragma unroll
      for (int nt = 0; nt < 4; nt++) mx = fmaxf(mx, S[mt][nt][e]);
#pragma unroll
      for (int msk = 1; msk < 16; msk <<= 1) mx = fmaxf(mx, __shfl_xor(mx, msk, 32));
      float sum = 0.f;
#pragma unroll
      for (int nt = 0; nt < 4; nt++) {
        float pe = __expf(S[mt][nt][e] - mx);
        S[mt][nt][e] = pe; sum += pe;
      }
#pragma unroll
      for (int msk = 1; msk < 16; msk <<= 1) sum += __shfl_xor(sum, msk, 32);
      float inv = 1.f / sum;
#pragma unroll
      for (int nt = 0; nt < 4; nt++) S[mt][nt][e] *= inv;
    }

  // ---- probs -> LDS (reuse own rpe slice; private per wave) for transpose
#pragma unroll
  for (int mt = 0; mt < 4; mt++)
#pragma unroll
    for (int nt = 0; nt < 4; nt++)
#pragma unroll
      for (int e = 0; e < 8; e++) {
        int r = mt * 16 + e + ((lane < 16) ? 0 : 8);
        int c = nt * 16 + (lane & 15);
        s_rpe[h * 4096 + r * 64 + c] = (f16)S[mt][nt][e];
      }

  // ---- phase 2: out = P @ V_eff  (64x64 @ 64x32)
  v8f o2[4][2];
  { v8f z = {};
#pragma unroll
    for (int mt = 0; mt < 4; mt++) { o2[mt][0] = z; o2[mt][1] = z; } }
#pragma unroll
  for (int k0 = 0; k0 < 64; k0 += 32) {
    v16h pa[4], vb[2];
#pragma unroll
    for (int mt = 0; mt < 4; mt++) {
      const f16* bp = s_rpe + h * 4096 + (mt * 16 + (lane & 15)) * 64 + k0 + ((lane < 16) ? 0 : 16);
#pragma unroll
      for (int t = 0; t < 16; t++) pa[mt][t] = bp[t];
    }
#pragma unroll
    for (int ni = 0; ni < 2; ni++) {
      const f16* bp = s_vt + h * 2304 + (ni * 16 + (lane & 15)) * 72 + k0 + ((lane < 16) ? 0 : 16);
#pragma unroll
      for (int t = 0; t < 16; t++) vb[ni][t] = bp[t];
    }
#pragma unroll
    for (int mt = 0; mt < 4; mt++)
#pragma unroll
      for (int ni = 0; ni < 2; ni++)
        o2[mt][ni] = __builtin_amdgcn_wmma_f32_16x16x32_f16(
            false, pa[mt], false, vb[ni], (short)0, o2[mt][ni], false, false);
  }

  // ---- store [row, h*32+d] f16
#pragma unroll
  for (int mt = 0; mt < 4; mt++)
#pragma unroll
    for (int ni = 0; ni < 2; ni++)
#pragma unroll
      for (int e = 0; e < 8; e++) {
        int r = mt * 16 + e + ((lane < 16) ? 0 : 8);
        if (r < 49) {
          int d = ni * 16 + (lane & 15);
          attn_out[(rowbase + r) * 256 + h * 32 + d] = (f16)o2[mt][ni][e];
        }
      }
}

// ---------------------------------------------------------------- launcher
extern "C" void kernel_launch(void* const* d_in, const int* in_sizes, int n_in,
                              void* d_out, int out_size, void* d_ws, size_t ws_size,
                              hipStream_t stream)
{
  const float* x       = (const float*)d_in[0];
  const float* maskm   = (const float*)d_in[1];
  const float* polar   = (const float*)d_in[2];
  const float* vote    = (const float*)d_in[3];
  const float* g1      = (const float*)d_in[4];
  const float* be1     = (const float*)d_in[5];
  const float* w_qkv   = (const float*)d_in[6];
  const float* b_qkv   = (const float*)d_in[7];
  const float* w_proj  = (const float*)d_in[8];
  const float* b_proj  = (const float*)d_in[9];
  const float* tau     = (const float*)d_in[10];
  const float* rpe_w1  = (const float*)d_in[11];
  const float* rpe_b1  = (const float*)d_in[12];
  const float* rpe_w2  = (const float*)d_in[13];
  const float* rpe_b2  = (const float*)d_in[14];
  const float* vote_w1 = (const float*)d_in[15];
  const float* vote_b1 = (const float*)d_in[16];
  const float* vote_w2 = (const float*)d_in[17];
  const float* vote_b2 = (const float*)d_in[18];
  const float* g2      = (const float*)d_in[19];
  const float* be2     = (const float*)d_in[20];
  const float* fc1_w   = (const float*)d_in[21];
  const float* fc1_b   = (const float*)d_in[22];
  const float* fc2_w   = (const float*)d_in[23];
  const float* fc2_b   = (const float*)d_in[24];
  float* out = (float*)d_out;

  char* ws = (char*)d_ws;
  size_t off = 0;
  auto alloc = [&](size_t bytes) -> char* {
    char* p = ws + off; off += (bytes + 255) & ~(size_t)255; return p;
  };
  f16*   wqkvh  = (f16*)alloc((size_t)768 * 256 * 2);
  f16*   wprojh = (f16*)alloc((size_t)256 * 256 * 2);
  f16*   wfc1h  = (f16*)alloc((size_t)1024 * 256 * 2);
  f16*   wfc2h  = (f16*)alloc((size_t)256 * 1024 * 2);
  f16*   xw     = (f16*)alloc((size_t)NTOK * 256 * 2);
  f16*   vlin   = (f16*)alloc((size_t)NTOK * 256 * 2);
  float* pw     = (float*)alloc((size_t)NTOK * 2 * 4);
  f16*   qkvh   = (f16*)alloc((size_t)NTOK * 768 * 2);
  float* x2     = (float*)alloc((size_t)NTOK * 256 * 4);
  f16*   h1     = (f16*)alloc((size_t)NTOK * 1024 * 2);
  f16*   attn_o = xw;    // xw dead after QKV gemm
  f16*   yh     = vlin;  // vlin dead after QKV gemm epilogue

  cvt_f32_f16<<<(768 * 256 + 255) / 256, 256, 0, stream>>>(w_qkv, wqkvh, 768 * 256);
  cvt_f32_f16<<<(256 * 256 + 255) / 256, 256, 0, stream>>>(w_proj, wprojh, 256 * 256);
  cvt_f32_f16<<<(1024 * 256 + 255) / 256, 256, 0, stream>>>(fc1_w, wfc1h, 1024 * 256);
  cvt_f32_f16<<<(256 * 1024 + 255) / 256, 256, 0, stream>>>(fc2_w, wfc2h, 256 * 1024);

  ln1_window_kernel<<<NTOK / 8, 256, 0, stream>>>(x, g1, be1, polar, xw, pw);
  vote_window_kernel<<<NTOK / 8, 256, 0, stream>>>(vote, vote_w1, vote_b1, vote_w2, vote_b2, vlin);

  GemmP p0{ xw, wqkvh, b_qkv, 256, 0, vlin, qkvh, nullptr, nullptr };
  gemm_wmma_kernel<<<dim3(NTOK / 128, 768 / 128), 256, 0, stream>>>(p0);

  attn_kernel<<<NWIN, 256, 0, stream>>>(qkvh, pw, maskm, tau,
                                        rpe_w1, rpe_b1, rpe_w2, rpe_b2, attn_o);

  GemmP p1{ attn_o, wprojh, b_proj, 256, 1, nullptr, nullptr, x, x2 };
  gemm_wmma_kernel<<<dim3(NTOK / 128, 256 / 128), 256, 0, stream>>>(p1);

  ln2_kernel<<<NTOK / 8, 256, 0, stream>>>(x2, g2, be2, yh);

  GemmP p2{ yh, wfc1h, fc1_b, 256, 2, nullptr, h1, nullptr, nullptr };
  gemm_wmma_kernel<<<dim3(NTOK / 128, 1024 / 128), 256, 0, stream>>>(p2);

  GemmP p3{ h1, wfc2h, fc2_b, 1024, 3, nullptr, nullptr, x2, out };
  gemm_wmma_kernel<<<dim3(NTOK / 128, 256 / 128), 256, 0, stream>>>(p3);
}
